// SelfAttention_86517821213276
// MI455X (gfx1250) — matrix operand order repaired and checked by reading, not compiled, not run on hardware
//
#include <hip/hip_runtime.h>

typedef _Float16 f16;
typedef __attribute__((ext_vector_type(16))) _Float16 v16h;
typedef __attribute__((ext_vector_type(8)))  float    v8f;

static constexpr int kB  = 8;
static constexpr int kC  = 256;   // channels
static constexpr int kN  = 4096;  // h*w
static constexpr int kNT = kN / 16;   // 256 column tiles of 16
static constexpr int kKT = kC / 32;   // 8 K-chunks of 32 for QKV gemm
static constexpr int kRT = 20;        // (32+32+256)/16 output-row tiles
static constexpr int kMC = kN / 32;   // 128 m-chunks of 32

// Workspace layout (units: f16 elements). Total ~37.9 MB.
static constexpr size_t QP = 0;                                   // [B][NT][32][16]   q A-fragments
static constexpr size_t KP = QP + (size_t)kB * kNT * 512;         // [B][NT][32][16]   k B-fragments
static constexpr size_t VP = KP + (size_t)kB * kNT * 512;         // [B][MC][16][32][16] v^T B-fragments
static constexpr size_t XP = VP + (size_t)kB * kMC * 16 * 512;    // [B][KT][NT][32][16] x B-fragments
static constexpr size_t WP = XP + (size_t)kB * kKT * kNT * 512;   // [RT][KT][32][16]  W A-fragments

__device__ __forceinline__ v8f wmma16(v16h a, v16h b, v8f c) {
    return __builtin_amdgcn_wmma_f32_16x16x32_f16(false, a, false, b, (short)0, c, false, false);
}

// CDNA5 async global->LDS copy (ASYNCcnt-tracked DMA; bypasses VGPRs).
// vdst VGPR carries the LDS byte address, vaddr the 64-bit global address.
__device__ __forceinline__ void async_copy_b128(unsigned lds_off, const f16* gptr) {
    asm volatile("global_load_async_to_lds_b128 %0, %1, off"
                 :: "v"(lds_off), "v"((unsigned long long)(uintptr_t)gptr)
                 : "memory");
}
__device__ __forceinline__ void wait_async0() {
    asm volatile("s_wait_asynccnt 0" ::: "memory");
}

// ---- A-fragment mapping (16x32 f16, M x K): element (r, kk) ----
// lanes 0-15 hold M=lane with K in {0..7, 16..23}; lanes 16-31 hold M=lane-16 with K in {8..15, 24..31}
__device__ __forceinline__ int laneA(int r, int kk) { return r + (((kk >> 3) & 1) << 4); }
__device__ __forceinline__ int elemA(int kk)        { return (kk & 7) + (((kk >> 4) & 1) << 3); }
// ---- B-fragment mapping (32x16 f16, K x N): element (kk, nn) ----
// lanes 0-15 hold K in {0..7, 16..23} of column nn; lanes 16-31 hold K in {8..15, 24..31}
__device__ __forceinline__ int laneB(int nn, int kk) { return nn + (((kk >> 3) & 1) << 4); }
__device__ __forceinline__ int elemB(int kk)         { return (kk & 7) + (((kk >> 4) & 1) << 3); }

// ------------------------------------------------------------------
// Kernel 0a: convert x (f32 [B][C][N]) into f16 B-fragments
// ------------------------------------------------------------------
__global__ __launch_bounds__(256) void pack_x_kernel(const float* __restrict__ x,
                                                     f16* __restrict__ xp) {
    int idx = blockIdx.x * 256 + threadIdx.x;          // over B*C*N, n fastest -> coalesced reads
    int n = idx & (kN - 1);
    int c = (idx >> 12) & (kC - 1);
    int b = idx >> 20;
    float v = x[idx];
    int kt = c >> 5, kk = c & 31;
    int nt = n >> 4, nn = n & 15;
    xp[((((size_t)b * kKT + kt) * kNT + nt) * 32 + laneB(nn, kk)) * 16 + elemB(kk)] = (f16)v;
}

// ------------------------------------------------------------------
// Kernel 0b: pack stacked weights [wq;wk;wv] (320x256) into A-fragments
// ------------------------------------------------------------------
__global__ __launch_bounds__(256) void pack_w_kernel(const float* __restrict__ wq,
                                                     const float* __restrict__ wk,
                                                     const float* __restrict__ wv,
                                                     f16* __restrict__ wp) {
    int idx = blockIdx.x * 256 + threadIdx.x;          // over 320*256
    int c = idx & 255;
    int row = idx >> 8;
    float v = (row < 32) ? wq[row * 256 + c]
            : (row < 64) ? wk[(row - 32) * 256 + c]
                         : wv[(row - 64) * 256 + c];
    int rt = row >> 4, rr = row & 15;
    int kt = c >> 5, kk = c & 31;
    wp[(((size_t)rt * kKT + kt) * 32 + laneA(rr, kk)) * 16 + elemA(kk)] = (f16)v;
}

// ------------------------------------------------------------------
// Kernel 1: QKV projection via WMMA.  One wave per (b, nt, rt) tile:
//   out[rt*16.., nt*16..] = W[320,256] x X[256,4096] + bias
// Results are scattered (element-wise f16 stores) into q/k/v fragment layouts.
// ------------------------------------------------------------------
__global__ __launch_bounds__(128) void qkv_kernel(const f16* __restrict__ xp,
                                                  const f16* __restrict__ wp,
                                                  const float* __restrict__ bq,
                                                  const float* __restrict__ bk,
                                                  const float* __restrict__ bv,
                                                  f16* __restrict__ qp,
                                                  f16* __restrict__ kp,
                                                  f16* __restrict__ vp) {
    const int lane = threadIdx.x & 31;
    const int wv32 = threadIdx.x >> 5;
    const int nt = blockIdx.x;
    const int rt = blockIdx.y * 4 + wv32;   // 0..19
    const int b  = blockIdx.z;

    v8f acc = {0.f, 0.f, 0.f, 0.f, 0.f, 0.f, 0.f, 0.f};
#pragma unroll
    for (int kt = 0; kt < kKT; ++kt) {
        v16h a  = *(const v16h*)(wp + (((size_t)rt * kKT + kt) * 32 + lane) * 16);
        v16h bb = *(const v16h*)(xp + ((((size_t)b * kKT + kt) * kNT + nt) * 32 + lane) * 16);
        acc = wmma16(a, bb, acc);
    }

    const int half = lane >> 4;
    const int lanelo = lane & 15;
#pragma unroll
    for (int j = 0; j < 8; ++j) {
        int row = rt * 16 + j + half * 8;   // output row in stacked [q;k;v]
        float bias = (row < 32) ? bq[row] : (row < 64) ? bk[row - 32] : bv[row - 64];
        f16 hv = (f16)(acc[j] + bias);
        if (row < 32) {             // q[d][n] -> A-fragment of q (M=n rows, K=d)
            int d = row;
            qp[(((size_t)b * kNT + nt) * 32 + laneA(lanelo, d)) * 16 + elemA(d)] = hv;
        } else if (row < 64) {      // k[d][m] -> B-fragment (K=d, N=m tile nt)
            int d = row - 32;
            kp[(((size_t)b * kNT + nt) * 32 + laneB(lanelo, d)) * 16 + elemB(d)] = hv;
        } else {                    // v[c][m] -> vT B-fragment (K=m chunk, N=c tile)
            int c = row - 64;
            int mc = nt >> 1;
            int kk = ((nt & 1) << 4) | lanelo;   // m % 32
            int ct = c >> 4, nn2 = c & 15;
            vp[((((size_t)b * kMC + mc) * 16 + ct) * 32 + laneB(nn2, kk)) * 16 + elemB(kk)] = hv;
        }
    }
}

// ------------------------------------------------------------------
// Kernel 2: flash-style attention. One wave owns a 16-row query block
// and the full 256-channel output (16 accumulator tiles). The 4 waves
// of a block share one batch, so k/v operand tiles are staged ONCE per
// block into LDS via async global->LDS DMA (ASYNCcnt), overlapped with
// the energy WMMAs / softmax work.
// ------------------------------------------------------------------
__global__ __launch_bounds__(128) void attn_kernel(const f16* __restrict__ qp,
                                                   const f16* __restrict__ kp,
                                                   const f16* __restrict__ vp,
                                                   const float* __restrict__ x,
                                                   const float* __restrict__ gamma,
                                                   float* __restrict__ out) {
    const int lane = threadIdx.x & 31;
    const int wv32 = threadIdx.x >> 5;
    const int half = lane >> 4;
    const int lanelo = lane & 15;
    const int nblk = blockIdx.x * 4 + wv32;   // 0..255 query-row tile
    const int b = blockIdx.y;

    __shared__ f16 ldsk[4 * 32 * 16];          // 4 staged k B-fragment tiles (4 KB)
    __shared__ f16 ldsv[16 * 32 * 16];         // one staged v chunk: 16 B-fragment tiles (16 KB)
    __shared__ f16 ldsp[4 * 16 * 48];          // per-wave 16x32 p-tile transpose (6 KB)

    // q A-fragment for this row block (reused across all m)
    const v16h aq = *(const v16h*)(qp + (((size_t)b * kNT + nblk) * 32 + lane) * 16);

    // ---------- pass 1: row max (k tiles staged 4-at-a-time via async DMA) ----------
    float rm[8];
#pragma unroll
    for (int j = 0; j < 8; ++j) rm[j] = -1e30f;
    const v8f zero = {0.f, 0.f, 0.f, 0.f, 0.f, 0.f, 0.f, 0.f};
    for (int mt4 = 0; mt4 < kNT; mt4 += 4) {
        {   // wave w stages tile mt4+w (1 KB): 2 async b128 per lane
            const f16* g = kp + (((size_t)b * kNT + (mt4 + wv32)) * 32 + lane) * 16;
            unsigned la = (unsigned)(uintptr_t)&ldsk[((size_t)wv32 * 32 + lane) * 16];
            async_copy_b128(la, g);
            async_copy_b128(la + 16, g + 8);
        }
        wait_async0();
        __syncthreads();
#pragma unroll
        for (int t = 0; t < 4; ++t) {
            v16h bk_ = *(const v16h*)&ldsk[((size_t)t * 32 + lane) * 16];
            v8f e = wmma16(aq, bk_, zero);
#pragma unroll
            for (int j = 0; j < 8; ++j) rm[j] = fmaxf(rm[j], e[j]);
        }
        __syncthreads();
    }
#pragma unroll
    for (int off = 1; off <= 8; off <<= 1)
#pragma unroll
        for (int j = 0; j < 8; ++j) rm[j] = fmaxf(rm[j], __shfl_xor(rm[j], off, 32));

    // ---------- pass 2: exp, row sums, out accumulation ----------
    float rs[8];
#pragma unroll
    for (int j = 0; j < 8; ++j) rs[j] = 0.f;
    v8f acc[16];
#pragma unroll
    for (int t = 0; t < 16; ++t) acc[t] = zero;

    for (int mc = 0; mc < kMC; ++mc) {
        // stage this chunk's 16 v tiles (16 KB); wave w loads tiles 4w..4w+3.
        // DMA runs concurrently with the energy/softmax work below.
#pragma unroll
        for (int t = 0; t < 4; ++t) {
            int ct = wv32 * 4 + t;
            const f16* g = vp + ((((size_t)b * kMC + mc) * 16 + ct) * 32 + lane) * 16;
            unsigned la = (unsigned)(uintptr_t)&ldsv[((size_t)ct * 32 + lane) * 16];
            async_copy_b128(la, g);
            async_copy_b128(la + 16, g + 8);
        }

        v16h b0 = *(const v16h*)(kp + (((size_t)b * kNT + 2 * mc + 0) * 32 + lane) * 16);
        v16h b1 = *(const v16h*)(kp + (((size_t)b * kNT + 2 * mc + 1) * 32 + lane) * 16);
        v8f e0 = wmma16(aq, b0, zero);
        v8f e1 = wmma16(aq, b1, zero);
#pragma unroll
        for (int j = 0; j < 8; ++j) {
            float p0 = __expf(e0[j] - rm[j]);
            float p1 = __expf(e1[j] - rm[j]);
            rs[j] += p0 + p1;
            int row = j + half * 8;                       // C-layout: lane holds one column
            ldsp[(wv32 * 16 + row) * 48 + lanelo]      = (f16)p0;   // cols 0..15 of chunk
            ldsp[(wv32 * 16 + row) * 48 + lanelo + 16] = (f16)p1;   // cols 16..31
        }
        wait_async0();
        __syncthreads();  // v chunk staged + p-tiles visible
        v16h ap;
        {
            const f16* rowp = &ldsp[(wv32 * 16 + lanelo) * 48];     // A-fragment row = lane&15
            int kb = half * 8;
#pragma unroll
            for (int e2 = 0; e2 < 8; ++e2) ap[e2] = rowp[kb + e2];          // K = kb..kb+7
#pragma unroll
            for (int e2 = 0; e2 < 8; ++e2) ap[8 + e2] = rowp[16 + kb + e2]; // K = 16+kb..
        }
#pragma unroll
        for (int ct = 0; ct < 16; ++ct) {
            v16h bv_ = *(const v16h*)&ldsv[((size_t)ct * 32 + lane) * 16];
            acc[ct] = wmma16(ap, bv_, acc[ct]);
        }
        __syncthreads();  // before next iteration overwrites ldsv/ldsp
    }

    // reduce row sums across the 16-lane half
#pragma unroll
    for (int off = 1; off <= 8; off <<= 1)
#pragma unroll
        for (int j = 0; j < 8; ++j) rs[j] += __shfl_xor(rs[j], off, 32);
    float inv[8];
#pragma unroll
    for (int j = 0; j < 8; ++j) inv[j] = 1.0f / rs[j];

    const float g = gamma[0];
#pragma unroll
    for (int ct = 0; ct < 16; ++ct) {
        int c = ct * 16 + lanelo;
        size_t base = (((size_t)b * kC + c) * kN) + (size_t)nblk * 16 + half * 8;
#pragma unroll
        for (int j = 0; j < 8; ++j) {   // rows are contiguous in n -> compiler merges to b128
            out[base + j] = g * acc[ct][j] * inv[j] + x[base + j];
        }
    }
}

extern "C" void kernel_launch(void* const* d_in, const int* in_sizes, int n_in,
                              void* d_out, int out_size, void* d_ws, size_t ws_size,
                              hipStream_t stream) {
    (void)in_sizes; (void)n_in; (void)out_size; (void)ws_size;
    const float* x  = (const float*)d_in[0];
    const float* wq = (const float*)d_in[1];
    const float* bq = (const float*)d_in[2];
    const float* wk = (const float*)d_in[3];
    const float* bk = (const float*)d_in[4];
    const float* wv = (const float*)d_in[5];
    const float* bv = (const float*)d_in[6];
    const float* gm = (const float*)d_in[7];
    float* out = (float*)d_out;
    f16* ws = (f16*)d_ws;

    // pack inputs to WMMA fragment layouts
    pack_x_kernel<<<(kB * kC * kN) / 256, 256, 0, stream>>>(x, ws + XP);
    pack_w_kernel<<<(kRT * 16 * kC) / 256, 256, 0, stream>>>(wq, wk, wv, ws + WP);
    // QKV projections (WMMA)
    qkv_kernel<<<dim3(kNT, 5, kB), 128, 0, stream>>>(ws + XP, ws + WP, bq, bk, bv,
                                                     ws + QP, ws + KP, ws + VP);
    // flash attention (WMMA + async LDS staging) + residual
    attn_kernel<<<dim3(kN / 64, kB), 128, 0, stream>>>(ws + QP, ws + KP, ws + VP, x, gm, out);
}
